// CaptioningRNN_13786845020298
// MI455X (gfx1250) — compile-verified, matrix operand order checked by reading, not run on hardware
//
#include <hip/hip_runtime.h>

// Problem constants
#define NB 256      // batch
#define TT 64       // time steps
#define DD 512      // input dim
#define HH 1024     // hidden dim
#define FH 4096     // 4*H
#define KC 2560     // D + H + H  (fused GEMM K)
#define KSTEPS 80   // KC / 32

// Packed-B layout strides (bytes)
#define B_KS_BYTES   1024            // 32 lanes * 16 bf16 * 2B per (ct,ks) block
#define B_TILE_BYTES (KSTEPS * B_KS_BYTES)   // 81920 per 16-column tile

typedef __attribute__((ext_vector_type(16))) __bf16 v16bf;
typedef __attribute__((ext_vector_type(8)))  __bf16 v8bf;
typedef __attribute__((ext_vector_type(8)))  float  v8f;

union FragBF { v16bf v; v8bf h[2]; };

static __device__ __forceinline__ v8f wmma_bf16(const FragBF& a, const FragBF& b,
                                                v8f c)
{
    return __builtin_amdgcn_wmma_f32_16x16x32_bf16(
        false, a.v, false, b.v, (short)0, c, false, false);
}

// ---------------------------------------------------------------------------
// One-time: pack [Wx; Wh; Wattn] (2560 x 4096, fp32 row-major) into bf16
// pre-swizzled B-fragment layout: per (column-tile ct, k-step ks), 32 lanes x
// 16 bf16 contiguous.  Lane l -> n = ct*16 + (l&15); kb = ks*32 + (l>=16 ? 8:0);
// dword j<4 -> K = kb+2j ; j>=4 -> K = kb+16+2(j-4)   (ISA 05_wmma 7.12.2)
// ---------------------------------------------------------------------------
__global__ __launch_bounds__(256)
void rnn_pack_weights(const float* __restrict__ Wx, const float* __restrict__ Wh,
                      const float* __restrict__ Wattn, __bf16* __restrict__ wB)
{
    size_t gid = (size_t)blockIdx.x * 256 + threadIdx.x;      // dword index
    const size_t total = (size_t)256 * KSTEPS * 32 * 8;       // 5,242,880
    if (gid >= total) return;
    int j    = (int)(gid & 7);
    int lane = (int)((gid >> 3) & 31);
    int rem  = (int)(gid >> 8);
    int ks   = rem % KSTEPS;
    int ct   = rem / KSTEPS;                                   // 0..255
    int nn   = ct * 16 + (lane & 15);
    int kb   = ks * 32 + ((lane >> 4) ? 8 : 0);
    int k    = (j < 4) ? (kb + 2 * j) : (kb + 16 + 2 * (j - 4));
    float v0, v1;
    if (k < DD)            { v0 = Wx[(size_t)k * FH + nn];
                             v1 = Wx[(size_t)(k + 1) * FH + nn]; }
    else if (k < DD + HH)  { v0 = Wh[(size_t)(k - DD) * FH + nn];
                             v1 = Wh[(size_t)(k - DD + 1) * FH + nn]; }
    else                   { v0 = Wattn[(size_t)(k - DD - HH) * FH + nn];
                             v1 = Wattn[(size_t)(k - DD - HH + 1) * FH + nn]; }
    size_t off = (((size_t)ct * KSTEPS + ks) * 32 + lane) * 16 + (size_t)j * 2;
    wB[off]     = (__bf16)v0;
    wB[off + 1] = (__bf16)v1;
}

// One-time: x -> bf16
__global__ __launch_bounds__(256)
void rnn_pack_x(const float* __restrict__ x, __bf16* __restrict__ xbf, int n)
{
    int i = blockIdx.x * 256 + threadIdx.x;
    if (i < n) xbf[i] = (__bf16)x[i];
}

// One-time: h0 = c0 = mean over 4x4 grid of A
__global__ __launch_bounds__(256)
void rnn_init_h0(const float* __restrict__ A, float* __restrict__ hbuf,
                 float* __restrict__ cbuf)
{
    int gid = blockIdx.x * 256 + threadIdx.x;          // N*H
    if (gid >= NB * HH) return;
    const float* ap = A + (size_t)gid * 16;
    float s = 0.f;
    #pragma unroll
    for (int p = 0; p < 16; ++p) s += ap[p];
    s *= (1.0f / 16.0f);
    hbuf[gid] = s;
    cbuf[gid] = s;
}

// ---------------------------------------------------------------------------
// Per step: attention over the 4x4 grid + pack bf16 activation rows
// act[n] = [ x[n,t,:] (512) | h_prev (1024) | attn (1024) ]  as bf16
// One workgroup (256 threads = 8 waves) per batch row n.
// ---------------------------------------------------------------------------
__global__ __launch_bounds__(256)
void rnn_attn_pack(const float* __restrict__ A, const float* __restrict__ hbuf,
                   const __bf16* __restrict__ xbf, __bf16* __restrict__ act,
                   int tstep)
{
    __shared__ float spart[8][16];
    const int n    = blockIdx.x;
    const int tid  = threadIdx.x;
    const int lane = tid & 31;
    const int wave = tid >> 5;

    float partial[16];
    #pragma unroll
    for (int p = 0; p < 16; ++p) partial[p] = 0.f;

    for (int h = tid; h < HH; h += 256) {
        float hv = hbuf[n * HH + h];
        const float* ap = A + ((size_t)n * HH + h) * 16;
        #pragma unroll
        for (int p = 0; p < 16; ++p) partial[p] += hv * ap[p];
    }
    // deterministic reduce: intra-wave shuffle, then fixed-order LDS sum
    #pragma unroll
    for (int off = 16; off >= 1; off >>= 1) {
        #pragma unroll
        for (int p = 0; p < 16; ++p)
            partial[p] += __shfl_xor(partial[p], off, 32);
    }
    if (lane == 0) {
        #pragma unroll
        for (int p = 0; p < 16; ++p) spart[wave][p] = partial[p];
    }
    __syncthreads();

    const float scale = 0.03125f;      // 1/sqrt(1024)
    float sc[16];
    float mx = -3.0e38f;
    #pragma unroll
    for (int p = 0; p < 16; ++p) {
        float s = 0.f;
        #pragma unroll
        for (int w = 0; w < 8; ++w) s += spart[w][p];
        s *= scale;
        sc[p] = s;
        mx = fmaxf(mx, s);
    }
    float sum = 0.f;
    #pragma unroll
    for (int p = 0; p < 16; ++p) { sc[p] = __expf(sc[p] - mx); sum += sc[p]; }
    float inv = 1.f / sum;
    #pragma unroll
    for (int p = 0; p < 16; ++p) sc[p] *= inv;

    __bf16* arow = act + (size_t)n * KC;
    for (int h = tid; h < HH; h += 256) {
        const float* ap = A + ((size_t)n * HH + h) * 16;
        float av = 0.f;
        #pragma unroll
        for (int p = 0; p < 16; ++p) av += ap[p] * sc[p];
        arow[DD + HH + h] = (__bf16)av;
        arow[DD + h]      = (__bf16)hbuf[n * HH + h];
    }
    for (int k = tid; k < DD; k += 256)
        arow[k] = xbf[((size_t)n * TT + tstep) * DD + k];
}

// ---------------------------------------------------------------------------
// Per step: fused GEMM (256x2560 @ 2560x4096, bf16 WMMA, fp32 acc) + LSTM cell.
// Grid 256 WGs: mb = batch 16-row block, jb = 64-wide gate-column block.
// Wave w (0..3) owns gate quadrant w (i/f/o/g) -> 4 WMMA tiles; accumulators
// exchanged via LDS so the nonlinearity fuses in-kernel.
// All fragment addresses are immediate offsets from two base pointers; the
// k-loop is unrolled x2 with all 20 b128 loads issued before the 8 WMMAs so
// the scheduler can overlap loads with matrix ops via partial loadcnt waits.
// ---------------------------------------------------------------------------
__global__ __launch_bounds__(128)
void rnn_step_gemm_lstm(const __bf16* __restrict__ act,  // N x KC bf16
                        const __bf16* __restrict__ wB,   // packed B-fragments
                        const float*  __restrict__ bias, // 4H
                        float* __restrict__ hbuf,
                        float* __restrict__ cbuf,
                        float* __restrict__ out,         // N x T x H
                        int tstep)
{
    __shared__ float lds[4][16][64];
    const int lane = threadIdx.x & 31;
    const int wave = threadIdx.x >> 5;
    const int mb   = blockIdx.x >> 4;     // 0..15
    const int jb   = blockIdx.x & 15;     // 0..15

    v8f acc0 = (v8f){0.f,0.f,0.f,0.f,0.f,0.f,0.f,0.f};
    v8f acc1 = acc0, acc2 = acc0, acc3 = acc0;

    const int m      = mb * 16 + (lane & 15);
    const int khalf  = (lane >> 4) * 8;
    const int ctbase = wave * 64 + jb * 4;     // global 16-col tile index base

    // Base pointers; everything else is a compile-time byte offset.
    const char* aptr = (const char*)act + ((size_t)m * KC + khalf) * 2;
    const char* bptr = (const char*)wB + (size_t)ctbase * B_TILE_BYTES
                                       + (size_t)lane * 32;

    for (int ks = 0; ks < KSTEPS; ks += 2) {
        FragBF a0, a1, b0[4], b1[4];
        // ---- issue all loads for two k-steps -----------------------------
        a0.h[0] = *(const v8bf*)(aptr + 0);
        a0.h[1] = *(const v8bf*)(aptr + 32);
        a1.h[0] = *(const v8bf*)(aptr + 64);
        a1.h[1] = *(const v8bf*)(aptr + 96);
        #pragma unroll
        for (int tl = 0; tl < 4; ++tl) {
            b0[tl].h[0] = *(const v8bf*)(bptr + tl * B_TILE_BYTES + 0);
            b0[tl].h[1] = *(const v8bf*)(bptr + tl * B_TILE_BYTES + 16);
            b1[tl].h[0] = *(const v8bf*)(bptr + tl * B_TILE_BYTES + B_KS_BYTES + 0);
            b1[tl].h[1] = *(const v8bf*)(bptr + tl * B_TILE_BYTES + B_KS_BYTES + 16);
        }
        __builtin_prefetch(aptr + 128, 0, 1);        // next A rows
        __builtin_prefetch(bptr + 2 * B_KS_BYTES, 0, 1); // next B blocks
        // ---- 8 WMMAs ------------------------------------------------------
        acc0 = wmma_bf16(a0, b0[0], acc0);
        acc1 = wmma_bf16(a0, b0[1], acc1);
        acc2 = wmma_bf16(a0, b0[2], acc2);
        acc3 = wmma_bf16(a0, b0[3], acc3);
        acc0 = wmma_bf16(a1, b1[0], acc0);
        acc1 = wmma_bf16(a1, b1[1], acc1);
        acc2 = wmma_bf16(a1, b1[2], acc2);
        acc3 = wmma_bf16(a1, b1[3], acc3);
        aptr += 128;                 // 2 k-steps * 64 B
        bptr += 2 * B_KS_BYTES;      // 2 k-steps * 1 KB
    }

    // D-fragment layout: element j of lane l -> row (j + 8*(l>>4)), col (l&15)
    const int rr = (lane >> 4) * 8;
    const int cc = lane & 15;
    #pragma unroll
    for (int j = 0; j < 8; ++j) {
        lds[wave][rr + j][ 0 + cc] = acc0[j];
        lds[wave][rr + j][16 + cc] = acc1[j];
        lds[wave][rr + j][32 + cc] = acc2[j];
        lds[wave][rr + j][48 + cc] = acc3[j];
    }
    __syncthreads();

    for (int idx = threadIdx.x; idx < 16 * 64; idx += 128) {
        int r  = idx >> 6;
        int jj = idx & 63;
        int jcol = jb * 64 + jj;
        float iv = lds[0][r][jj] + bias[jcol];
        float fv = lds[1][r][jj] + bias[HH + jcol];
        float ov = lds[2][r][jj] + bias[2 * HH + jcol];
        float gv = lds[3][r][jj] + bias[3 * HH + jcol];
        int n = mb * 16 + r;
        size_t hidx = (size_t)n * HH + jcol;
        float cold = cbuf[hidx];
        float si = 1.f / (1.f + __expf(-iv));
        float sf = 1.f / (1.f + __expf(-fv));
        float so = 1.f / (1.f + __expf(-ov));
        float tg = tanhf(gv);
        float cn = sf * cold + si * tg;
        float hn = so * tanhf(cn);
        cbuf[hidx] = cn;
        hbuf[hidx] = hn;
        out[((size_t)n * TT + tstep) * HH + jcol] = hn;
    }
}

// ---------------------------------------------------------------------------
extern "C" void kernel_launch(void* const* d_in, const int* in_sizes, int n_in,
                              void* d_out, int out_size, void* d_ws, size_t ws_size,
                              hipStream_t stream)
{
    const float* x     = (const float*)d_in[0];   // N,T,D
    const float* A     = (const float*)d_in[1];   // N,H,16
    const float* Wx    = (const float*)d_in[2];   // D,4H
    const float* Wh    = (const float*)d_in[3];   // H,4H
    const float* Wattn = (const float*)d_in[4];   // H,4H
    const float* bias  = (const float*)d_in[5];   // 4H
    float* out = (float*)d_out;                   // N,T,H

    char* ws = (char*)d_ws;
    __bf16* wB   = (__bf16*)(ws);                 // 20,971,520 B packed weights
    __bf16* xbf  = (__bf16*)(ws + 20971520);      // 16,777,216 B bf16 x
    __bf16* act  = (__bf16*)(ws + 37748736);      //  1,310,720 B activation rows
    float*  hbuf = (float*) (ws + 39059456);      //  1,048,576 B h state
    float*  cbuf = (float*) (ws + 40108032);      //  1,048,576 B c state

    // one-time packing / init
    rnn_pack_weights<<<(int)(((size_t)256 * KSTEPS * 32 * 8 + 255) / 256), 256, 0,
                       stream>>>(Wx, Wh, Wattn, wB);
    {
        int nx = NB * TT * DD;
        rnn_pack_x<<<(nx + 255) / 256, 256, 0, stream>>>(x, xbf, nx);
    }
    rnn_init_h0<<<(NB * HH + 255) / 256, 256, 0, stream>>>(A, hbuf, cbuf);

    // recurrent steps
    for (int t = 0; t < TT; ++t) {
        rnn_attn_pack<<<NB, 256, 0, stream>>>(A, hbuf, xbf, act, t);
        rnn_step_gemm_lstm<<<256, 128, 0, stream>>>(act, wB, bias, hbuf, cbuf,
                                                    out, t);
    }
}
